// FlashMHA_76845554860298
// MI455X (gfx1250) — matrix operand order repaired and checked by reading, not compiled, not run on hardware
//
#include <hip/hip_runtime.h>
#include <hip/hip_bf16.h>
#include <math.h>

#define DIMC   2048
#define HEADS  16
#define HDIM   128
#define BATCH  2
#define SEQ    2048
#define HD     (HEADS*HDIM)    // 2048
#define MTOT   (BATCH*SEQ)     // 4096

typedef __attribute__((ext_vector_type(16))) __bf16 v16bf;
typedef __attribute__((ext_vector_type(8)))  float  v8f;

union FragU { v16bf v; uint4 q[2]; unsigned short s[16]; };

__device__ __forceinline__ unsigned short f2bf(float f) {
  union { float f; unsigned u; } c; c.f = f;
  unsigned r = c.u + 0x7fffu + ((c.u >> 16) & 1u);
  return (unsigned short)(r >> 16);
}

// A fragment: 16x32 bf16, row-major source (K contiguous), lda in elements.
__device__ __forceinline__ v16bf loadA(const __bf16* base, int lda, int row0, int col0, int lane) {
  int m = lane & 15, hf = lane >> 4;
  const __bf16* p = base + (size_t)(row0 + m) * lda + col0;
  FragU f;
  f.q[0] = *(const uint4*)(p + hf * 16);
  f.q[1] = *(const uint4*)(p + hf * 16 + 8);
  return f.v;
}

// B fragment: 32x16 bf16 from column-major storage Bt[n][k] (K contiguous), ldb in elements.
__device__ __forceinline__ v16bf loadB(const __bf16* bt, int ldb, int k0, int n0, int lane) {
  int n = lane & 15, hf = lane >> 4;
  const __bf16* p = bt + (size_t)(n0 + n) * ldb + k0 + hf * 16;
  FragU f;
  f.q[0] = *(const uint4*)(p);
  f.q[1] = *(const uint4*)(p + 8);
  return f.v;
}

__device__ __forceinline__ v8f wmma_bf16(v16bf a, v16bf b, v8f c) {
  return __builtin_amdgcn_wmma_f32_16x16x32_bf16(false, a, false, b, (short)0, c, false, false);
}

__device__ __forceinline__ float rowmax16(float v) {
  #pragma unroll
  for (int m = 1; m < 16; m <<= 1) v = fmaxf(v, __shfl_xor(v, m, 32));
  return v;
}
__device__ __forceinline__ float rowsum16(float v) {
  #pragma unroll
  for (int m = 1; m < 16; m <<= 1) v += __shfl_xor(v, m, 32);
  return v;
}

// ---------------- conversion kernels ----------------
// vectorized: one float4 -> 4 bf16 (8B store) per thread
__global__ void cvt_bf16_kernel(const float* __restrict__ in, __bf16* __restrict__ out, int n4) {
  int i = blockIdx.x * blockDim.x + threadIdx.x;
  if (i < n4) {
    float4 v = ((const float4*)in)[i];
    union { unsigned short s[4]; uint2 u; } p;
    p.s[0] = f2bf(v.x); p.s[1] = f2bf(v.y); p.s[2] = f2bf(v.z); p.s[3] = f2bf(v.w);
    ((uint2*)out)[i] = p.u;
  }
}

// LDS-tiled transpose: in K x N row-major f32 -> out N x K (K contiguous) bf16
// grid = (N/32, K/32), block = 256
__global__ __launch_bounds__(256)
void transpose_bf16_kernel(const float* __restrict__ in, __bf16* __restrict__ out,
                           int K, int N) {
  __shared__ unsigned short tile[32][34];
  int tx = threadIdx.x & 31;
  int ty = threadIdx.x >> 5;          // 0..7
  int n0 = blockIdx.x * 32;
  int k0 = blockIdx.y * 32;
  #pragma unroll
  for (int s = 0; s < 32; s += 8)
    tile[ty + s][tx] = f2bf(in[(size_t)(k0 + ty + s) * N + n0 + tx]);
  __syncthreads();
  #pragma unroll
  for (int s = 0; s < 32; s += 8)
    ((unsigned short*)out)[(size_t)(n0 + ty + s) * K + k0 + tx] = tile[tx][ty + s];
}

// ---------------- fused QKV projection + RoPE (64x64 per wave) ----------------
// grid = (MTOT/64, HD/64, 3), block = 32
__global__ __launch_bounds__(32)
void qkv_rope_kernel(const __bf16* __restrict__ xb,
                     const __bf16* __restrict__ WqT, const __bf16* __restrict__ WkT,
                     const __bf16* __restrict__ WvT,
                     const float* __restrict__ bq, const float* __restrict__ bk,
                     const float* __restrict__ bv,
                     __bf16* __restrict__ qout, __bf16* __restrict__ kout,
                     __bf16* __restrict__ vTout) {
  int lane = threadIdx.x & 31;
  int hf = lane >> 4, nloc = lane & 15;
  int row0 = blockIdx.x * 64, n0 = blockIdx.y * 64;
  int z = blockIdx.z;

  const __bf16* wt = (z == 0) ? WqT : (z == 1) ? WkT : WvT;
  const float* bias = (z == 0) ? bq : (z == 1) ? bk : bv;

  v8f acc[4][4];
  #pragma unroll
  for (int i = 0; i < 4; i++)
    #pragma unroll
    for (int j = 0; j < 4; j++) acc[i][j] = {};

  for (int k0 = 0; k0 < DIMC; k0 += 32) {
    v16bf a[4], b[4];
    #pragma unroll
    for (int i = 0; i < 4; i++) a[i] = loadA(xb, DIMC, row0 + 16 * i, k0, lane);
    #pragma unroll
    for (int j = 0; j < 4; j++) b[j] = loadB(wt, DIMC, k0, n0 + 16 * j, lane);
    #pragma unroll
    for (int i = 0; i < 4; i++)
      #pragma unroll
      for (int j = 0; j < 4; j++)
        acc[i][j] = wmma_bf16(a[i], b[j], acc[i][j]);
  }

  #pragma unroll
  for (int j = 0; j < 4; j++) {
    int n = n0 + 16 * j + nloc;
    int hidx = n >> 7;
    int d = n & 127;
    float bval = bias[n];
    float invf = exp2f(-((float)(d & ~1) / 128.0f) * 13.287712379549449f);
    #pragma unroll
    for (int i = 0; i < 4; i++) {
      #pragma unroll
      for (int r = 0; r < 8; r++) {
        int gm = row0 + 16 * i + r + 8 * hf;
        int bIdx = gm >> 11;
        int t = gm & (SEQ - 1);
        float val = acc[i][j][r] + bval;
        if (z < 2) {
          float prt = __shfl_xor(val, 1, 32);
          float sn, cs;
          __sincosf((float)t * invf, &sn, &cs);
          val = ((d & 1) == 0) ? (val * cs - prt * sn) : (prt * sn + val * cs);
        }
        unsigned short bfv = f2bf(val);
        size_t hb = (size_t)(bIdx * HEADS + hidx);
        if (z == 0)      ((unsigned short*)qout)[(hb * SEQ + t) * HDIM + d] = bfv;
        else if (z == 1) ((unsigned short*)kout)[(hb * SEQ + t) * HDIM + d] = bfv;
        else             ((unsigned short*)vTout)[(hb * HDIM + d) * SEQ + t] = bfv;
      }
    }
  }
}

// ---------------- flash attention ----------------
// grid = (SEQ/64, BATCH*HEADS), block = 128 (4 waves; 64 q-rows per block share K/V via LDS)
#define KPAD 136   // 128 + 8 elements: row stride 272B -> conflict-free b128 reads
#define VPAD 40    // 32 + 8 elements: row stride 80B  -> conflict-free b128 reads
__global__ __launch_bounds__(128)
void attn_kernel(const __bf16* __restrict__ qb, const __bf16* __restrict__ kb,
                 const __bf16* __restrict__ vT, __bf16* __restrict__ aout) {
  __shared__ __bf16 kls[32][KPAD];      // [key][d]
  __shared__ __bf16 vls[HDIM][VPAD];    // [d][key]
  __shared__ float  pls[4][16][33];     // per-wave P staging

  int tid  = threadIdx.x;
  int w    = tid >> 5;
  int lane = tid & 31;
  int hf = lane >> 4, nloc = lane & 15;
  int t0 = blockIdx.x * 64 + w * 16;
  int bh = blockIdx.y;

  const __bf16* qh = qb + (size_t)bh * SEQ * HDIM;
  const __bf16* kh = kb + (size_t)bh * SEQ * HDIM;
  const __bf16* vh = vT + (size_t)bh * HDIM * SEQ;

  v16bf aq[4];
  #pragma unroll
  for (int kk = 0; kk < 4; kk++) aq[kk] = loadA(qh, HDIM, t0, kk * 32, lane);

  v8f o[8];
  #pragma unroll
  for (int j = 0; j < 8; j++) o[j] = {};
  float mrow[8], lrow[8];
  #pragma unroll
  for (int r = 0; r < 8; r++) { mrow[r] = -1e30f; lrow[r] = 0.0f; }
  const float scale = 0.08838834764831845f;   // 1/sqrt(128)

  for (int kt = 0; kt < SEQ; kt += 32) {
    // cooperative global -> LDS staging of K (32x128) and V^T (128x32) tiles
    #pragma unroll
    for (int rep = 0; rep < 4; rep++) {
      int idx = tid + rep * 128;            // 512 x 16B chunks per tile
      int key = idx >> 4, dcol = (idx & 15) * 8;
      *(uint4*)&kls[key][dcol] = *(const uint4*)(kh + (size_t)(kt + key) * HDIM + dcol);
      int dr = idx >> 2, koff = (idx & 3) * 8;
      *(uint4*)&vls[dr][koff] = *(const uint4*)(vh + (size_t)dr * SEQ + kt + koff);
    }
    __syncthreads();

    // S = Q @ K^T : two 16x16 score tiles from LDS K
    v8f s0 = {}, s1 = {};
    #pragma unroll
    for (int kk = 0; kk < 4; kk++) {
      FragU fb0, fb1;
      const __bf16* p0 = &kls[nloc][kk * 32 + hf * 16];
      const __bf16* p1 = &kls[16 + nloc][kk * 32 + hf * 16];
      fb0.q[0] = *(const uint4*)(p0); fb0.q[1] = *(const uint4*)(p0 + 8);
      fb1.q[0] = *(const uint4*)(p1); fb1.q[1] = *(const uint4*)(p1 + 8);
      s0 = wmma_bf16(aq[kk], fb0.v, s0);
      s1 = wmma_bf16(aq[kk], fb1.v, s1);
    }

    // online softmax (row stats replicated across 16-lane half-groups)
    float fr[8];
    #pragma unroll
    for (int r = 0; r < 8; r++) {
      float a0 = s0[r] * scale, a1 = s1[r] * scale;
      float rm = rowmax16(fmaxf(a0, a1));
      float nm = fmaxf(mrow[r], rm);
      float f  = __expf(mrow[r] - nm);
      float p0 = __expf(a0 - nm), p1 = __expf(a1 - nm);
      lrow[r] = lrow[r] * f + rowsum16(p0 + p1);
      mrow[r] = nm;
      fr[r] = f;
      pls[w][r + 8 * hf][nloc] = p0;
      pls[w][r + 8 * hf][16 + nloc] = p1;
    }
    __syncthreads();

    // re-gather P as 16x32 bf16 A fragment
    FragU pf;
    #pragma unroll
    for (int j = 0; j < 16; j++) {
      int key = hf * 16 + j;
      pf.s[j] = f2bf(pls[w][nloc][key]);
    }

    // O = P @ V from LDS V^T
    #pragma unroll
    for (int j = 0; j < 8; j++) {
      FragU fv;
      const __bf16* pv = &vls[j * 16 + nloc][hf * 16];
      fv.q[0] = *(const uint4*)(pv); fv.q[1] = *(const uint4*)(pv + 8);
      v8f t = o[j];
      #pragma unroll
      for (int r = 0; r < 8; r++) t[r] *= fr[r];
      o[j] = wmma_bf16(pf.v, fv.v, t);
    }
    __syncthreads();
  }

  // epilogue: normalize and store bf16 [b, t, h*128+d]
  int bIdx = bh / HEADS, h = bh % HEADS;
  #pragma unroll
  for (int r = 0; r < 8; r++) {
    float inv = 1.0f / lrow[r];
    int t = t0 + r + 8 * hf;
    size_t rowbase = ((size_t)(bIdx * SEQ + t)) * HD + h * HDIM;
    #pragma unroll
    for (int j = 0; j < 8; j++)
      ((unsigned short*)aout)[rowbase + j * 16 + nloc] = f2bf(o[j][r] * inv);
  }
}

// ---------------- output projection (64x64 per wave) ----------------
// grid = (MTOT/64, DIMC/64), block = 32
__global__ __launch_bounds__(32)
void oproj_kernel(const __bf16* __restrict__ ab, const __bf16* __restrict__ WoT,
                  const float* __restrict__ bo, float* __restrict__ out) {
  int lane = threadIdx.x & 31;
  int hf = lane >> 4, nloc = lane & 15;
  int row0 = blockIdx.x * 64, n0 = blockIdx.y * 64;

  v8f acc[4][4];
  #pragma unroll
  for (int i = 0; i < 4; i++)
    #pragma unroll
    for (int j = 0; j < 4; j++) acc[i][j] = {};

  for (int k0 = 0; k0 < HD; k0 += 32) {
    v16bf a[4], b[4];
    #pragma unroll
    for (int i = 0; i < 4; i++) a[i] = loadA(ab, HD, row0 + 16 * i, k0, lane);
    #pragma unroll
    for (int j = 0; j < 4; j++) b[j] = loadB(WoT, HD, k0, n0 + 16 * j, lane);
    #pragma unroll
    for (int i = 0; i < 4; i++)
      #pragma unroll
      for (int j = 0; j < 4; j++)
        acc[i][j] = wmma_bf16(a[i], b[j], acc[i][j]);
  }

  #pragma unroll
  for (int j = 0; j < 4; j++) {
    float bias = bo[n0 + 16 * j + nloc];
    #pragma unroll
    for (int i = 0; i < 4; i++)
      #pragma unroll
      for (int r = 0; r < 8; r++)
        out[(size_t)(row0 + 16 * i + r + 8 * hf) * DIMC + n0 + 16 * j + nloc] =
            acc[i][j][r] + bias;
  }
}

extern "C" void kernel_launch(void* const* d_in, const int* in_sizes, int n_in,
                              void* d_out, int out_size, void* d_ws, size_t ws_size,
                              hipStream_t stream) {
  (void)in_sizes; (void)n_in; (void)out_size; (void)ws_size;
  const float* x  = (const float*)d_in[0];
  const float* Wq = (const float*)d_in[1];
  const float* bq = (const float*)d_in[2];
  const float* Wk = (const float*)d_in[3];
  const float* bk = (const float*)d_in[4];
  const float* Wv = (const float*)d_in[5];
  const float* bv = (const float*)d_in[6];
  const float* Wo = (const float*)d_in[7];
  const float* bo = (const float*)d_in[8];
  float* out = (float*)d_out;

  char* ws = (char*)d_ws;
  size_t off = 0;
  auto alloc = [&](size_t bytes) { void* p = ws + off; off += (bytes + 255) & ~(size_t)255; return p; };

  __bf16* xb  = (__bf16*)alloc((size_t)MTOT * DIMC * 2);
  __bf16* WqT = (__bf16*)alloc((size_t)DIMC * HD * 2);
  __bf16* WkT = (__bf16*)alloc((size_t)DIMC * HD * 2);
  __bf16* WvT = (__bf16*)alloc((size_t)DIMC * HD * 2);
  __bf16* WoT = (__bf16*)alloc((size_t)HD * DIMC * 2);
  __bf16* qb  = (__bf16*)alloc((size_t)BATCH * HEADS * SEQ * HDIM * 2);
  __bf16* kb  = (__bf16*)alloc((size_t)BATCH * HEADS * SEQ * HDIM * 2);
  __bf16* vT  = (__bf16*)alloc((size_t)BATCH * HEADS * HDIM * SEQ * 2);
  __bf16* ab  = (__bf16*)alloc((size_t)MTOT * HD * 2);

  int n4 = MTOT * DIMC / 4;
  cvt_bf16_kernel<<<(n4 + 255) / 256, 256, 0, stream>>>(x, xb, n4);
  dim3 gt(HD / 32, DIMC / 32);
  transpose_bf16_kernel<<<gt, 256, 0, stream>>>(Wq, WqT, DIMC, HD);
  transpose_bf16_kernel<<<gt, 256, 0, stream>>>(Wk, WkT, DIMC, HD);
  transpose_bf16_kernel<<<gt, 256, 0, stream>>>(Wv, WvT, DIMC, HD);
  dim3 gto(DIMC / 32, HD / 32);
  transpose_bf16_kernel<<<gto, 256, 0, stream>>>(Wo, WoT, HD, DIMC);

  dim3 g2(MTOT / 64, HD / 64, 3);
  qkv_rope_kernel<<<g2, 32, 0, stream>>>(xb, WqT, WkT, WvT, bq, bk, bv, qb, kb, vT);

  dim3 g3(SEQ / 64, BATCH * HEADS);
  attn_kernel<<<g3, 128, 0, stream>>>(qb, kb, vT, ab);

  dim3 g4(MTOT / 64, DIMC / 64);
  oproj_kernel<<<g4, 32, 0, stream>>>(ab, WoT, bo, out);
}
